// ECGFM_67765993996577
// MI455X (gfx1250) — matrix operand order repaired and checked by reading, not compiled, not run on hardware
//
#include <hip/hip_runtime.h>

// ---------------- model constants (from reference) ----------------
#define CTOK   12
#define BATCH  32
#define LSEQ   600
#define TWIN   50
#define DMODEL 768
#define DMLP   3072
#define DDEC   512
#define ENC_DEP 12
#define DEC_DEP 4
#define SENC   132          // 12 cls + 120 kept
#define SDEC   612          // 12 cls + 600
#define LKEEP  120          // int(600 * 0.2)
#define NPL    50           // L / C
#define TOKE   (BATCH * SENC)   // 4224  = 33 * 128
#define TOKD   (BATCH * SDEC)   // 19584 = 153 * 128
#define SPAD_E 160          // SENC padded to 32
#define SPAD_D 640          // SDEC padded to 32
#define SLD_E  164          // LDS row stride (bank-conflict pad)
#define SLD_D  644

// ---------------- WMMA plumbing ----------------
typedef __bf16 bf16_t;
typedef __attribute__((ext_vector_type(16))) __bf16 v16bf;
typedef __attribute__((ext_vector_type(8)))  float  v8f;

__device__ inline v8f wmma_bf16(v16bf a, v16bf b, v8f c) {
  // D = A(16x32 bf16) * B(32x16 bf16) + C(16x16 f32)
  return __builtin_amdgcn_wmma_f32_16x16x32_bf16(false, a, false, b, (short)0, c,
                                                 false, false);
}

__device__ inline unsigned pack_bf16_2(float a, float b) {
  union { bf16_t h[2]; unsigned u; } x;
  x.h[0] = (bf16_t)a;
  x.h[1] = (bf16_t)b;
  return x.u;   // lowers to v_cvt_pk_bf16_f32
}

// A fragment 16x32 (M x K), row-major LDS tile with leading dim ld.
// lane<16: K 0..15 ; lane>=16: K 16..31 ; M = lane&15
__device__ inline v16bf load_a_frag(const bf16_t* p, int ld, int lane) {
  const int m = lane & 15;
  const int kb = (lane >> 4) * 16;
  const bf16_t* row = p + m * ld;
  v16bf a;
#pragma unroll
  for (int v = 0; v < 8; v++) {
    int k = kb + 2 * v;
    a[2 * v]     = row[k];
    a[2 * v + 1] = row[k + 1];
  }
  return a;
}

// B fragment 32x16 (K x N), row-major LDS tile with leading dim ld.
// lane<16: K 0..15 ; lane>=16: K 16..31 ; N = lane&15
__device__ inline v16bf load_b_frag(const bf16_t* p, int ld, int lane) {
  const int n = lane & 15;
  const int kb = (lane >> 4) * 16;
  v16bf b;
#pragma unroll
  for (int v = 0; v < 8; v++) {
    int k = kb + 2 * v;
    b[2 * v]     = p[k * ld + n];
    b[2 * v + 1] = p[(k + 1) * ld + n];
  }
  return b;
}

__device__ inline float gelu_f(float x) {
  float x3 = x * x * x;
  return 0.5f * x * (1.0f + tanhf(0.7978845608028654f * (x + 0.044715f * x3)));
}

// ======== FAST GEMM (guard-free): M%128==0, N%256==0, K%32==0 ========
// 256 threads / 8 waves, block tile 128x256, wave tile 64x64 (4x4 frags),
// float4 global loads -> packed bf16 -> ds_store_b64, next-tile prefetch.
#define ALD 40   // padded A leading dim (bank-conflict-free b128 reads)
__global__ __launch_bounds__(256) void gemm_bf16_fast_kernel(
    const float* __restrict__ A, const float* __restrict__ W,
    const float* __restrict__ bias, const float* __restrict__ residual,
    float* __restrict__ Cout, int M, int N, int K, int act)
{
  __shared__ bf16_t lA[128 * ALD];   // 10 KB
  __shared__ bf16_t lB[32 * 256];    // 16 KB
  const int tid  = threadIdx.x;
  const int lane = tid & 31;
  const int wave = tid >> 5;
  const int wm = wave >> 2;        // 0..1 : 64-row wave tile
  const int wn = wave & 3;         // 0..3 : 64-col wave tile
  const int bm = blockIdx.y * 128;
  const int bn = blockIdx.x * 256;

  v8f acc[4][4];
#pragma unroll
  for (int i = 0; i < 4; i++)
#pragma unroll
    for (int j = 0; j < 4; j++)
#pragma unroll
      for (int v = 0; v < 8; v++) acc[i][j][v] = 0.0f;

  for (int k0 = 0; k0 < K; k0 += 32) {
    // ---- A tile 128x32 : 1024 float4, 4 per thread ----
#pragma unroll
    for (int i = 0; i < 4; i++) {
      int idx = tid + i * 256;
      int e = idx * 4;
      int r = e >> 5, c = e & 31;
      const float4 v = *(const float4*)(A + (size_t)(bm + r) * K + k0 + c);
      unsigned p0 = pack_bf16_2(v.x, v.y);
      unsigned p1 = pack_bf16_2(v.z, v.w);
      uint2 pk; pk.x = p0; pk.y = p1;
      *(uint2*)(lA + r * ALD + c) = pk;
    }
    // ---- B tile 32x256 : 2048 float4, 8 per thread ----
#pragma unroll
    for (int i = 0; i < 8; i++) {
      int idx = tid + i * 256;
      int e = idx * 4;
      int r = e >> 8, c = e & 255;
      const float4 v = *(const float4*)(W + (size_t)(k0 + r) * N + bn + c);
      unsigned p0 = pack_bf16_2(v.x, v.y);
      unsigned p1 = pack_bf16_2(v.z, v.w);
      uint2 pk; pk.x = p0; pk.y = p1;
      *(uint2*)(lB + r * 256 + c) = pk;
    }
    if (k0 + 32 < K) {   // prefetch next K tile (global_prefetch_b8)
      __builtin_prefetch(A + (size_t)(bm + (tid >> 1)) * K + k0 + 32 + (tid & 1) * 16, 0, 1);
      __builtin_prefetch(W + (size_t)(k0 + 32 + (tid >> 3)) * N + bn + (tid & 7) * 32, 0, 1);
    }
    __syncthreads();
    v16bf af[4], bfm[4];
#pragma unroll
    for (int i = 0; i < 4; i++)
      af[i] = load_a_frag(lA + (wm * 64 + i * 16) * ALD, ALD, lane);
#pragma unroll
    for (int j = 0; j < 4; j++)
      bfm[j] = load_b_frag(lB + wn * 64 + j * 16, 256, lane);
#pragma unroll
    for (int i = 0; i < 4; i++)
#pragma unroll
      for (int j = 0; j < 4; j++)
        acc[i][j] = wmma_bf16(af[i], bfm[j], acc[i][j]);
    __syncthreads();
  }

  const int rb = (lane >> 4) * 8;
  const int cl = lane & 15;
#pragma unroll
  for (int i = 0; i < 4; i++) {
#pragma unroll
    for (int j = 0; j < 4; j++) {
      int col = bn + wn * 64 + j * 16 + cl;
      float bv = bias ? bias[col] : 0.0f;
#pragma unroll
      for (int v = 0; v < 8; v++) {
        int row = bm + wm * 64 + i * 16 + rb + v;
        float val = acc[i][j][v] + bv;
        if (act == 1) val = gelu_f(val);
        if (residual) val += residual[(size_t)row * N + col];
        Cout[(size_t)row * N + col] = val;
      }
    }
  }
}

// ======== EDGE GEMM (fully guarded), used only for N=50 pred head ========
__global__ __launch_bounds__(256) void gemm_bf16_kernel(
    const float* __restrict__ A, const float* __restrict__ W,
    const float* __restrict__ bias, const float* __restrict__ residual,
    float* __restrict__ Cout, int M, int N, int K, int act)
{
  __shared__ bf16_t lA[128 * 32];
  __shared__ bf16_t lB[32 * 128];
  const int tid  = threadIdx.x;
  const int lane = tid & 31;
  const int wave = tid >> 5;
  const int wm = wave >> 1;
  const int wn = wave & 1;
  const int bm = blockIdx.y * 128;
  const int bn = blockIdx.x * 128;

  v8f acc[2][4];
#pragma unroll
  for (int i = 0; i < 2; i++)
#pragma unroll
    for (int j = 0; j < 4; j++)
#pragma unroll
      for (int v = 0; v < 8; v++) acc[i][j][v] = 0.0f;

  for (int k0 = 0; k0 < K; k0 += 32) {
#pragma unroll
    for (int i = 0; i < 16; i++) {
      int e = tid + i * 256;
      int r = e >> 5, c = e & 31;
      int gr = bm + r, gc = k0 + c;
      float v = (gr < M && gc < K) ? A[(size_t)gr * K + gc] : 0.0f;
      lA[e] = (bf16_t)v;
    }
#pragma unroll
    for (int i = 0; i < 16; i++) {
      int e = tid + i * 256;
      int r = e >> 7, c = e & 127;
      int gr = k0 + r, gc = bn + c;
      float v = (gr < K && gc < N) ? W[(size_t)gr * N + gc] : 0.0f;
      lB[e] = (bf16_t)v;
    }
    __syncthreads();
    v16bf af[2], bfm[4];
#pragma unroll
    for (int i = 0; i < 2; i++)
      af[i] = load_a_frag(lA + (wm * 32 + i * 16) * 32, 32, lane);
#pragma unroll
    for (int j = 0; j < 4; j++)
      bfm[j] = load_b_frag(lB + wn * 64 + j * 16, 128, lane);
#pragma unroll
    for (int i = 0; i < 2; i++)
#pragma unroll
      for (int j = 0; j < 4; j++)
        acc[i][j] = wmma_bf16(af[i], bfm[j], acc[i][j]);
    __syncthreads();
  }

  const int rb = (lane >> 4) * 8;
  const int cl = lane & 15;
#pragma unroll
  for (int i = 0; i < 2; i++) {
#pragma unroll
    for (int j = 0; j < 4; j++) {
      int col = bn + wn * 64 + j * 16 + cl;
      if (col >= N) continue;
      float bv = bias ? bias[col] : 0.0f;
#pragma unroll
      for (int v = 0; v < 8; v++) {
        int row = bm + wm * 32 + i * 16 + rb + v;
        if (row >= M) continue;
        float val = acc[i][j][v] + bv;
        if (act == 1) val = gelu_f(val);
        if (residual) val += residual[(size_t)row * N + col];
        Cout[(size_t)row * N + col] = val;
      }
    }
  }
}

// ---------------- fused attention ----------------
// grid (ceil(S/64), H, B), block 128 (4 waves). Each wave: 16 q-rows.
// Dynamic LDS: 4 * 16 * Sld floats (per-wave score rows, padded stride).
template <int HD>
__global__ __launch_bounds__(128) void attn_kernel(
    const float* __restrict__ qkv, const unsigned char* __restrict__ blocked,
    float* __restrict__ o, int S, int Spad, int Sld, int Dm, float scale)
{
  extern __shared__ float smem[];
  const int lane = threadIdx.x & 31;
  const int wave = threadIdx.x >> 5;
  const int b = blockIdx.z, h = blockIdx.y;
  const int qbase = blockIdx.x * 64 + wave * 16;
  const int D3 = 3 * Dm;
  float* P = smem + (size_t)wave * 16 * Sld;
  for (int i = lane; i < 16 * Sld; i += 32) P[i] = 0.0f;

  const int m   = lane & 15;
  const int kbA = (lane >> 4) * 16;
  const int nI  = lane & 15;
  const int kbB = (lane >> 4) * 16;
  constexpr int NKF = HD / 32;   // Q frags over head dim
  constexpr int NOF = HD / 16;   // output col frags

  // Q fragments, loaded once and reused over all key tiles
  v16bf aq[NKF];
  if (qbase + 16 <= S) {          // uniform fast path: tile fully in-bounds
    const float* qp = qkv + (size_t)(b * S + qbase + m) * D3 + h * HD;
#pragma unroll
    for (int f = 0; f < NKF; f++)
#pragma unroll
      for (int v = 0; v < 8; v++) {
        int k = kbA + 2 * v;
        aq[f][2 * v]     = (bf16_t)qp[f * 32 + k];
        aq[f][2 * v + 1] = (bf16_t)qp[f * 32 + k + 1];
      }
  } else {
    int qrow = qbase + m;
    const float* qp = (qrow < S) ? (qkv + (size_t)(b * S + qrow) * D3 + h * HD)
                                 : (const float*)nullptr;
#pragma unroll
    for (int f = 0; f < NKF; f++)
#pragma unroll
      for (int v = 0; v < 8; v++) {
        int k = kbA + 2 * v;
        aq[f][2 * v]     = qp ? (bf16_t)qp[f * 32 + k]     : (bf16_t)0.0f;
        aq[f][2 * v + 1] = qp ? (bf16_t)qp[f * 32 + k + 1] : (bf16_t)0.0f;
      }
  }

  // S = Q K^T, scaled + masked, into LDS rows
  for (int kt = 0; kt < S; kt += 16) {
    v8f c;
#pragma unroll
    for (int v = 0; v < 8; v++) c[v] = 0.0f;
    int key = kt + nI;
    if (kt + 16 <= S) {           // uniform fast path
      const float* kp = qkv + (size_t)(b * S + key) * D3 + Dm + h * HD;
#pragma unroll
      for (int f = 0; f < NKF; f++) {
        v16bf bk;
#pragma unroll
        for (int v = 0; v < 8; v++) {
          int k = kbB + 2 * v;
          bk[2 * v]     = (bf16_t)kp[f * 32 + k];
          bk[2 * v + 1] = (bf16_t)kp[f * 32 + k + 1];
        }
        c = wmma_bf16(aq[f], bk, c);
      }
    } else {
      const float* kp = (key < S) ? (qkv + (size_t)(b * S + key) * D3 + Dm + h * HD)
                                  : (const float*)nullptr;
#pragma unroll
      for (int f = 0; f < NKF; f++) {
        v16bf bk;
#pragma unroll
        for (int v = 0; v < 8; v++) {
          int k = kbB + 2 * v;
          bk[2 * v]     = kp ? (bf16_t)kp[f * 32 + k]     : (bf16_t)0.0f;
          bk[2 * v + 1] = kp ? (bf16_t)kp[f * 32 + k + 1] : (bf16_t)0.0f;
        }
        c = wmma_bf16(aq[f], bk, c);
      }
    }
    if (key < S) {
      int rb = (lane >> 4) * 8;
#pragma unroll
      for (int v = 0; v < 8; v++) {
        int r = rb + v;
        int qrow = qbase + r;
        float val = c[v] * scale;
        if (blocked) {
          int qr = (qrow < S) ? qrow : (S - 1);
          if (blocked[(size_t)qr * S + key]) val = -1.0e9f;
        }
        P[r * Sld + key] = val;
      }
    }
  }
  asm volatile("s_wait_dscnt 0x0" ::: "memory");

  // row softmax (wave-cooperative, shuffle reductions)
  for (int r = 0; r < 16; r++) {
    float* row = P + r * Sld;
    float mx = -1.0e30f;
    for (int cdx = lane; cdx < S; cdx += 32) mx = fmaxf(mx, row[cdx]);
#pragma unroll
    for (int off = 16; off > 0; off >>= 1) mx = fmaxf(mx, __shfl_xor(mx, off, 32));
    float sum = 0.0f;
    for (int cdx = lane; cdx < S; cdx += 32) {
      float e = __expf(row[cdx] - mx);
      row[cdx] = e;
      sum += e;
    }
#pragma unroll
    for (int off = 16; off > 0; off >>= 1) sum += __shfl_xor(sum, off, 32);
    float inv = 1.0f / sum;
    for (int cdx = lane; cdx < S; cdx += 32) row[cdx] *= inv;
  }
  asm volatile("s_wait_dscnt 0x0" ::: "memory");

  // O = P V
  v8f accv[NOF];
#pragma unroll
  for (int f = 0; f < NOF; f++)
#pragma unroll
    for (int v = 0; v < 8; v++) accv[f][v] = 0.0f;

  for (int kt = 0; kt < Spad; kt += 32) {
    v16bf ap;
#pragma unroll
    for (int v = 0; v < 8; v++) {
      int k = kbA + 2 * v;
      ap[2 * v]     = (bf16_t)P[m * Sld + kt + k];
      ap[2 * v + 1] = (bf16_t)P[m * Sld + kt + k + 1];
    }
    if (kt + 32 <= S) {           // uniform fast path
#pragma unroll
      for (int f = 0; f < NOF; f++) {
        v16bf bv;
#pragma unroll
        for (int v = 0; v < 8; v++) {
          int k = kbB + 2 * v;
          int dcol = h * HD + f * 16 + nI;
          const float* vp = qkv + (size_t)(b * S + kt + k) * D3 + 2 * Dm + dcol;
          bv[2 * v]     = (bf16_t)vp[0];
          bv[2 * v + 1] = (bf16_t)vp[(size_t)D3];
        }
        accv[f] = wmma_bf16(ap, bv, accv[f]);
      }
    } else {
#pragma unroll
      for (int f = 0; f < NOF; f++) {
        v16bf bv;
#pragma unroll
        for (int v = 0; v < 8; v++) {
          int k = kbB + 2 * v;
          int s0 = kt + k, s1 = kt + k + 1;
          int dcol = h * HD + f * 16 + nI;
          float x0 = (s0 < S) ? qkv[(size_t)(b * S + s0) * D3 + 2 * Dm + dcol] : 0.0f;
          float x1 = (s1 < S) ? qkv[(size_t)(b * S + s1) * D3 + 2 * Dm + dcol] : 0.0f;
          bv[2 * v]     = (bf16_t)x0;
          bv[2 * v + 1] = (bf16_t)x1;
        }
        accv[f] = wmma_bf16(ap, bv, accv[f]);
      }
    }
  }
  int rb2 = (lane >> 4) * 8;
#pragma unroll
  for (int f = 0; f < NOF; f++) {
    int dcol = h * HD + f * 16 + nI;
#pragma unroll
    for (int v = 0; v < 8; v++) {
      int qrow = qbase + rb2 + v;
      if (qrow < S) o[(size_t)(b * S + qrow) * Dm + dcol] = accv[f][v];
    }
  }
}

// ---------------- LayerNorm (one block per token) ----------------
__global__ __launch_bounds__(256) void ln_kernel(const float* __restrict__ x,
                                                 const float* __restrict__ w,
                                                 const float* __restrict__ b,
                                                 float* __restrict__ y, int D)
{
  const int t = blockIdx.x;
  const float* xr = x + (size_t)t * D;
  float s = 0.0f, s2 = 0.0f;
  for (int i = threadIdx.x; i < D; i += 256) {
    float v = xr[i];
    s += v; s2 += v * v;
  }
  __shared__ float red[2][8];
#pragma unroll
  for (int off = 16; off > 0; off >>= 1) {
    s  += __shfl_xor(s, off, 32);
    s2 += __shfl_xor(s2, off, 32);
  }
  int lane = threadIdx.x & 31, wv = threadIdx.x >> 5;
  if (lane == 0) { red[0][wv] = s; red[1][wv] = s2; }
  __syncthreads();
  if (wv == 0) {
    s  = (lane < 8) ? red[0][lane] : 0.0f;
    s2 = (lane < 8) ? red[1][lane] : 0.0f;
#pragma unroll
    for (int off = 4; off > 0; off >>= 1) {
      s  += __shfl_xor(s, off, 32);
      s2 += __shfl_xor(s2, off, 32);
    }
    if (lane == 0) { red[0][0] = s; red[1][0] = s2; }
  }
  __syncthreads();
  float mean = red[0][0] / D;
  float var  = red[1][0] / D - mean * mean;
  float rstd = rsqrtf(var + 1e-5f);
  float* yr = y + (size_t)t * D;
  for (int i = threadIdx.x; i < D; i += 256)
    yr[i] = (xr[i] - mean) * rstd * w[i] + b[i];
}

// ---------------- signal normalization: x0 = (sig-mu)/sqrt(var_ddof1+1e-6) ----
__global__ __launch_bounds__(32) void prep_kernel(const float* __restrict__ sig,
                                                  float* __restrict__ x0)
{
  const int bl = blockIdx.x;
  const float* r = sig + (size_t)bl * TWIN;
  const int lane = threadIdx.x;
  float s = 0.0f, s2 = 0.0f;
  for (int i = lane; i < TWIN; i += 32) { float v = r[i]; s += v; s2 += v * v; }
#pragma unroll
  for (int off = 16; off > 0; off >>= 1) {
    s  += __shfl_xor(s, off, 32);
    s2 += __shfl_xor(s2, off, 32);
  }
  float mean = s / TWIN;
  float var  = (s2 - TWIN * mean * mean) / (TWIN - 1);
  float rstd = rsqrtf(var + 1e-6f);
  float* w = x0 + (size_t)bl * TWIN;
  for (int i = lane; i < TWIN; i += 32) w[i] = (r[i] - mean) * rstd;
}

// ---------------- argsort ranks of noise (stable), keep list, mask m -------
__global__ void rank_kernel(const float* __restrict__ noise,
                            int* __restrict__ restore, int* __restrict__ keep,
                            float* __restrict__ mfl)
{
  __shared__ float ln_[LSEQ];
  for (int i = threadIdx.x; i < LSEQ; i += blockDim.x) ln_[i] = noise[i];
  __syncthreads();
  for (int i = threadIdx.x; i < LSEQ; i += blockDim.x) {
    float vi = ln_[i];
    int r = 0;
    for (int j = 0; j < LSEQ; j++) {
      float vj = ln_[j];
      r += (vj < vi) || (vj == vi && j < i);
    }
    restore[i] = r;
    mfl[i] = (r >= LKEEP) ? 1.0f : 0.0f;
    if (r < LKEEP) keep[r] = i;
  }
}

// ---------------- wrap-conv patch embed + tables + pos ----------------
__global__ __launch_bounds__(256) void conv_embed_kernel(
    const float* __restrict__ x0, const float* __restrict__ conv_w,
    const float* __restrict__ spa, const float* __restrict__ tem,
    const float* __restrict__ pos, const int* __restrict__ chan,
    const int* __restrict__ timei, float* __restrict__ out)
{
  const int bl = blockIdx.x;
  const int b = bl / LSEQ, l = bl % LSEQ;
  __shared__ float ls[3 * TWIN];
  const int lm1 = (l + LSEQ - 1) % LSEQ, lp1 = (l + 1) % LSEQ;
  const float* r0 = x0 + ((size_t)b * LSEQ + lm1) * TWIN;
  const float* r1 = x0 + ((size_t)b * LSEQ + l)   * TWIN;
  const float* r2 = x0 + ((size_t)b * LSEQ + lp1) * TWIN;
  for (int i = threadIdx.x; i < TWIN; i += 256) {
    ls[i] = r0[i]; ls[TWIN + i] = r1[i]; ls[2 * TWIN + i] = r2[i];
  }
  __syncthreads();
  const int ci = chan[bl], ti = timei[bl];
  for (int d = threadIdx.x; d < DMODEL; d += 256) {
    const float* w = conv_w + (size_t)d * TWIN * 3;
    float acc = 0.0f;
    for (int tw = 0; tw < TWIN; tw++) {
      acc += w[tw * 3 + 0] * ls[tw]
           + w[tw * 3 + 1] * ls[TWIN + tw]
           + w[tw * 3 + 2] * ls[2 * TWIN + tw];
    }
    out[((size_t)b * LSEQ + l) * DMODEL + d] =
        acc + spa[(size_t)ci * DMODEL + d] + tem[(size_t)ti * DMODEL + d]
            + pos[(size_t)(CTOK + l) * DMODEL + d];
  }
}

// ---------------- build encoder input xe = [cls+pos ; gathered kept] -------
__global__ void build_xe_kernel(const float* __restrict__ emb,
                                const float* __restrict__ cls,
                                const float* __restrict__ pos,
                                const int* __restrict__ keep,
                                float* __restrict__ xe)
{
  const int ti = blockIdx.x;
  const int b = ti / SENC, t = ti % SENC;
  float* dst = xe + ((size_t)b * SENC + t) * DMODEL;
  if (t < CTOK) {
    for (int d = threadIdx.x; d < DMODEL; d += blockDim.x)
      dst[d] = cls[(size_t)t * DMODEL + d] + pos[(size_t)t * DMODEL + d];
  } else {
    int src = keep[t - CTOK];
    const float* s = emb + ((size_t)b * LSEQ + src) * DMODEL;
    for (int d = threadIdx.x; d < DMODEL; d += blockDim.x) dst[d] = s[d];
  }
}

// ---------------- restore/scatter decoder input xd ----------------
__global__ void build_xd_kernel(const float* __restrict__ xde,
                                const float* __restrict__ mtok,
                                const float* __restrict__ dpos,
                                const int* __restrict__ restore,
                                float* __restrict__ xd)
{
  const int ti = blockIdx.x;
  const int b = ti / SDEC, t = ti % SDEC;
  float* dst = xd + ((size_t)b * SDEC + t) * DDEC;
  for (int d = threadIdx.x; d < DDEC; d += blockDim.x) {
    float v;
    if (t < CTOK) {
      v = xde[((size_t)b * SENC + t) * DDEC + d];
    } else {
      int r = restore[t - CTOK];
      v = (r < LKEEP) ? xde[((size_t)b * SENC + CTOK + r) * DDEC + d] : mtok[d];
    }
    dst[d] = v + dpos[(size_t)t * DDEC + d];
  }
}

// ---------------- block-sparse decoder mask ----------------
__global__ void build_mask_kernel(const int* __restrict__ restore,
                                  unsigned char* __restrict__ blocked)
{
  const int idx = blockIdx.x * blockDim.x + threadIdx.x;
  if (idx >= SDEC * SDEC) return;
  const int i = idx / SDEC, j = idx % SDEC;
  bool allow;
  if (i < CTOK && j < CTOK) {
    allow = (i == j);
  } else if (i < CTOK) {
    allow = (i == (j - CTOK) / NPL);
  } else if (j < CTOK) {
    allow = (j == (i - CTOK) / NPL);
  } else {
    const int li = i - CTOK, lj = j - CTOK;
    const bool mi = restore[li] >= LKEEP;
    const bool mj = restore[lj] >= LKEEP;
    allow = mi ? ((li % NPL) == (lj % NPL)) : (!mj);
  }
  blocked[idx] = allow ? 0 : 1;
}

// ---------------- loss ----------------
__global__ void loss_zero_kernel(float* sc) {
  if (threadIdx.x == 0) { sc[0] = 0.0f; sc[1] = 0.0f; }
}

__global__ __launch_bounds__(32) void loss_kernel(const float* __restrict__ pred,
                                                  const float* __restrict__ tgt,
                                                  const float* __restrict__ mfl,
                                                  float* __restrict__ sc)
{
  const int bl = blockIdx.x;
  const int b = bl / LSEQ, l = bl % LSEQ;
  const float mv = mfl[l];
  const float* p = pred + ((size_t)b * SDEC + CTOK + l) * TWIN;
  const float* t = tgt + ((size_t)b * LSEQ + l) * TWIN;
  float s = 0.0f;
  for (int i = threadIdx.x; i < TWIN; i += 32) {
    float d = p[i] - t[i];
    s += d * d;
  }
#pragma unroll
  for (int off = 16; off > 0; off >>= 1) s += __shfl_xor(s, off, 32);
  if (threadIdx.x == 0) {
    atomicAdd(&sc[0], mv * s / (float)TWIN);
    if (b == 0) atomicAdd(&sc[1], mv);
  }
}

__global__ void loss_final_kernel(const float* sc, float* out) {
  if (threadIdx.x == 0) out[0] = sc[0] / sc[1];
}

// ---------------- host orchestration ----------------
extern "C" void kernel_launch(void* const* d_in, const int* in_sizes, int n_in,
                              void* d_out, int out_size, void* d_ws, size_t ws_size,
                              hipStream_t stream)
{
  (void)in_sizes; (void)n_in; (void)out_size; (void)ws_size;
  const float* signals = (const float*)d_in[0];
  const float* noise   = (const float*)d_in[1];
  const int*   chan    = (const int*)d_in[2];
  const int*   timei   = (const int*)d_in[3];
  const float* conv_w  = (const float*)d_in[4];
  const float* spa     = (const float*)d_in[5];
  const float* tem     = (const float*)d_in[6];
  const float* pos     = (const float*)d_in[7];
  const float* cls     = (const float*)d_in[8];
  const float* eWqkv = (const float*)d_in[9];
  const float* ebqkv = (const float*)d_in[10];
  const float* eWo   = (const float*)d_in[11];
  const float* ebo   = (const float*)d_in[12];
  const float* eln1w = (const float*)d_in[13];
  const float* eln1b = (const float*)d_in[14];
  const float* eln2w = (const float*)d_in[15];
  const float* eln2b = (const float*)d_in[16];
  const float* eW1   = (const float*)d_in[17];
  const float* eb1   = (const float*)d_in[18];
  const float* eW2   = (const float*)d_in[19];
  const float* eb2   = (const float*)d_in[20];
  const float* normw = (const float*)d_in[21];
  const float* normb = (const float*)d_in[22];
  const float* dembW = (const float*)d_in[23];
  const float* dembB = (const float*)d_in[24];
  const float* mtok  = (const float*)d_in[25];
  const float* dpos  = (const float*)d_in[26];
  const float* dWqkv = (const float*)d_in[27];
  const float* dbqkv = (const float*)d_in[28];
  const float* dWo   = (const float*)d_in[29];
  const float* dbo   = (const float*)d_in[30];
  const float* dln1w = (const float*)d_in[31];
  const float* dln1b = (const float*)d_in[32];
  const float* dln2w = (const float*)d_in[33];
  const float* dln2b = (const float*)d_in[34];
  const float* dW1   = (const float*)d_in[35];
  const float* db1   = (const float*)d_in[36];
  const float* dW2   = (const float*)d_in[37];
  const float* db2   = (const float*)d_in[38];
  const float* dnw   = (const float*)d_in[39];
  const float* dnb   = (const float*)d_in[40];
  const float* predW = (const float*)d_in[41];
  const float* predB = (const float*)d_in[42];

  // workspace layout (BIG is overlaid: conv-embed output, then MLP hidden)
  char* base = (char*)d_ws;
  size_t off = 0;
  auto alloc = [&](size_t bytes) -> char* {
    char* p = base + off;
    off += (bytes + 255) & ~(size_t)255;
    return p;
  };
  float* tgt  = (float*)alloc((size_t)BATCH * LSEQ * TWIN * 4);
  float* big  = (float*)alloc((size_t)TOKD * DMLP * 4);   // 240 MB
  float* emb  = big;                                      // (B,L,768) pre-encoder
  float* h1   = big;                                      // MLP hidden in blocks
  float* xe   = (float*)alloc((size_t)TOKE * DMODEL * 4);
  float* xd   = (float*)alloc((size_t)TOKD * DDEC * 4);
  float* hbuf = (float*)alloc((size_t)TOKD * DDEC * 4);   // >= TOKE*768
  float* qkv  = (float*)alloc((size_t)TOKD * 3 * DDEC * 4); // >= TOKE*2304
  float* obuf = (float*)alloc((size_t)TOKD * DDEC * 4);   // attn out / xde / pred
  int*   restore = (int*)alloc(LSEQ * 4);
  int*   keep    = (int*)alloc(128 * 4);
  float* mfl     = (float*)alloc(LSEQ * 4);
  unsigned char* blocked = (unsigned char*)alloc((size_t)SDEC * SDEC);
  float* sc = (float*)alloc(256);

  auto gemm = [&](const float* A, const float* W, const float* bi,
                  const float* res, float* Cp, int M, int N, int K, int act) {
    if ((M % 128 == 0) && (N % 256 == 0) && (K % 32 == 0)) {
      dim3 g(N / 256, M / 128);
      gemm_bf16_fast_kernel<<<g, 256, 0, stream>>>(A, W, bi, res, Cp, M, N, K, act);
    } else {
      dim3 g((N + 127) / 128, (M + 127) / 128);
      gemm_bf16_kernel<<<g, 256, 0, stream>>>(A, W, bi, res, Cp, M, N, K, act);
    }
  };

  // ---- preprocessing ----
  prep_kernel<<<BATCH * LSEQ, 32, 0, stream>>>(signals, tgt);
  rank_kernel<<<1, 256, 0, stream>>>(noise, restore, keep, mfl);
  conv_embed_kernel<<<BATCH * LSEQ, 256, 0, stream>>>(tgt, conv_w, spa, tem, pos,
                                                      chan, timei, emb);
  build_xe_kernel<<<BATCH * SENC, 256, 0, stream>>>(emb, cls, pos, keep, xe);

  // ---- encoder: 12 layers, D=768, 12 heads x 64 ----
  for (int i = 0; i < ENC_DEP; i++) {
    const float* Wq = eWqkv + (size_t)i * DMODEL * 3 * DMODEL;
    const float* bq = ebqkv + (size_t)i * 3 * DMODEL;
    const float* Wo = eWo + (size_t)i * DMODEL * DMODEL;
    const float* bo = ebo + (size_t)i * DMODEL;
    ln_kernel<<<TOKE, 256, 0, stream>>>(xe, eln1w + i * DMODEL, eln1b + i * DMODEL,
                                        hbuf, DMODEL);
    gemm(hbuf, Wq, bq, nullptr, qkv, TOKE, 3 * DMODEL, DMODEL, 0);
    attn_kernel<64><<<dim3((SENC + 63) / 64, 12, BATCH), 128,
                      4 * 16 * SLD_E * sizeof(float), stream>>>(
        qkv, nullptr, obuf, SENC, SPAD_E, SLD_E, DMODEL, 0.125f);
    gemm(obuf, Wo, bo, xe, xe, TOKE, DMODEL, DMODEL, 0);
    ln_kernel<<<TOKE, 256, 0, stream>>>(xe, eln2w + i * DMODEL, eln2b + i * DMODEL,
                                        hbuf, DMODEL);
    gemm(hbuf, eW1 + (size_t)i * DMODEL * DMLP, eb1 + (size_t)i * DMLP, nullptr,
         h1, TOKE, DMLP, DMODEL, 1);
    gemm(h1, eW2 + (size_t)i * DMLP * DMODEL, eb2 + (size_t)i * DMODEL, xe, xe,
         TOKE, DMODEL, DMLP, 0);
  }
  ln_kernel<<<TOKE, 256, 0, stream>>>(xe, normw, normb, hbuf, DMODEL);
  gemm(hbuf, dembW, dembB, nullptr, obuf /*xde*/, TOKE, DDEC, DMODEL, 0);

  build_xd_kernel<<<BATCH * SDEC, 256, 0, stream>>>(obuf, mtok, dpos, restore, xd);
  build_mask_kernel<<<(SDEC * SDEC + 255) / 256, 256, 0, stream>>>(restore, blocked);

  // ---- decoder: 4 layers, D=512, 16 heads x 32, block-sparse mask ----
  for (int i = 0; i < DEC_DEP; i++) {
    const float* Wq = dWqkv + (size_t)i * DDEC * 3 * DDEC;
    const float* bq = dbqkv + (size_t)i * 3 * DDEC;
    const float* Wo = dWo + (size_t)i * DDEC * DDEC;
    const float* bo = dbo + (size_t)i * DDEC;
    ln_kernel<<<TOKD, 256, 0, stream>>>(xd, dln1w + i * DDEC, dln1b + i * DDEC,
                                        hbuf, DDEC);
    gemm(hbuf, Wq, bq, nullptr, qkv, TOKD, 3 * DDEC, DDEC, 0);
    attn_kernel<32><<<dim3((SDEC + 63) / 64, 16, BATCH), 128,
                      4 * 16 * SLD_D * sizeof(float), stream>>>(
        qkv, blocked, obuf, SDEC, SPAD_D, SLD_D, DDEC, 0.17677669529663687f);
    gemm(obuf, Wo, bo, xd, xd, TOKD, DDEC, DDEC, 0);
    ln_kernel<<<TOKD, 256, 0, stream>>>(xd, dln2w + i * DDEC, dln2b + i * DDEC,
                                        hbuf, DDEC);
    gemm(hbuf, dW1 + (size_t)i * DDEC * DMLP, db1 + (size_t)i * DMLP, nullptr,
         h1, TOKD, DMLP, DDEC, 1);
    gemm(h1, dW2 + (size_t)i * DMLP * DDEC, db2 + (size_t)i * DDEC, xd, xd,
         TOKD, DDEC, DMLP, 0);
  }
  ln_kernel<<<TOKD, 256, 0, stream>>>(xd, dnw, dnb, hbuf, DDEC);
  gemm(hbuf, predW, predB, nullptr, obuf /*pred*/, TOKD, TWIN, DDEC, 0);

  loss_zero_kernel<<<1, 32, 0, stream>>>(sc);
  loss_kernel<<<BATCH * LSEQ, 32, 0, stream>>>(obuf, tgt, mfl, sc);
  loss_final_kernel<<<1, 32, 0, stream>>>(sc, (float*)d_out);
}
